// PetalExpertUnit_69922067579582
// MI455X (gfx1250) — matrix operand order repaired and checked by reading, not compiled, not run on hardware
//
#include <hip/hip_runtime.h>

// ---------------- problem constants ----------------
#define L_SEQ   4096     // 16*16*16 spatial positions
#define C_IN    128      // IN_DIM / OUT_DIM
#define NPART   4
#define DM      32       // D_MODEL
#define DI      64       // D_INNER
#define DSTATE  16
#define NINST   4        // (b=0,shared),(b=0,expert),(b=1,shared),(b=1,expert)
#define DBC_STRIDE 48    // 34 cols padded to 48 for 16-wide WMMA tiles

// ---------------- workspace layout (float offsets) ----------------
#define XN_OFF   0u                     // 4*4096*128      = 2,097,152
#define XZ_OFF   2097152u               // 16*4096*128     = 8,388,608
#define XI_OFF   10485760u              // 16*4096*64      = 4,194,304
#define DBC_OFF  14680064u              // 16*4096*48      = 3,145,728
#define DEL_OFF  17825792u              // 16*4096*64      = 4,194,304
#define Y_OFF    22020096u              // 16*4096*64      = 4,194,304
// total = 26,214,400 floats = 100 MB
// after the scan, xz region is dead -> alias later buffers into it:
#define XM_OFF   XZ_OFF                 // 4*4096*128
#define XM2_OFF  (XZ_OFF + 2097152u)    // 4*4096*128
#define PJ_OFF   (XZ_OFF + 4194304u)    // 4*4096*128

typedef __attribute__((ext_vector_type(16))) _Float16 v16h;
typedef __attribute__((ext_vector_type(8)))  float    v8f;

// mixer index: shared inst -> 0, expert inst -> 1 + min(dataset_id[b], 1)
__device__ __forceinline__ int mixer_of(int inst, const int* did) {
  if ((inst & 1) == 0) return 0;
  int id = did[inst >> 1];
  return 1 + (id < 1 ? id : 1);
}

__device__ __forceinline__ v8f wmma_f16(v16h a, v16h b, v8f c) {
  // D = A(16x32,f16) * B(32x16,f16) + C(16x16,f32)
  return __builtin_amdgcn_wmma_f32_16x16x32_f16(false, a, false, b, (short)0, c,
                                                false, false);
}

// A fragment: 16x32 tile, fp32 row-major (stride lda), converted to f16.
// ISA 7.12.2 16-bit A layout: lane<16 -> M=lane, K in {0..7,16..23};
// lane>=16 -> M=lane-16, K in {8..15,24..31}; VGPR v holds K pair.
__device__ __forceinline__ v16h load_a16x32(const float* A, int lda, int lane) {
  int r  = lane & 15;
  int kb = (lane >> 4) << 3;
  const float* row = A + (size_t)r * lda + kb;
  v16h a;
#pragma unroll
  for (int v = 0; v < 8; ++v) {
    int kk = 2 * v + ((v >= 4) ? 8 : 0);
    a[2 * v]     = (_Float16)row[kk];
    a[2 * v + 1] = (_Float16)row[kk + 1];
  }
  return a;
}

// B fragment: 32x16 tile of W^T where W is (N,K) fp32 row-major (stride ldw).
// lane<16 -> col=lane, K in {0..7,16..23}; lane>=16 -> col=lane-16, K in {8..15,24..31}.
// Rows of W beyond nvalid are zero-filled (for the 34-row x_proj weight).
__device__ __forceinline__ v16h load_b16x32(const float* W, int ldw, int nvalid,
                                            int lane) {
  int c  = lane & 15;
  int kb = (lane >> 4) << 3;
  bool ok = c < nvalid;
  int cc = ok ? c : 0;
  const float* row = W + (size_t)cc * ldw + kb;
  v16h b;
#pragma unroll
  for (int v = 0; v < 8; ++v) {
    int kk = 2 * v + ((v >= 4) ? 8 : 0);
    b[2 * v]     = ok ? (_Float16)row[kk]     : (_Float16)0.f;
    b[2 * v + 1] = ok ? (_Float16)row[kk + 1] : (_Float16)0.f;
  }
  return b;
}

// ---------------- kernel 1: layernorm of x (B,C,L) -> xn (inst,L,C) ----------
__global__ void k_ln1(const float* __restrict__ x, const float* __restrict__ ln_g,
                      const float* __restrict__ ln_b, const int* __restrict__ did,
                      float* __restrict__ ws) {
  int wid  = (blockIdx.x << 3) + (threadIdx.x >> 5);
  int lane = threadIdx.x & 31;
  int l    = wid & (L_SEQ - 1);
  int inst = wid >> 12;
  int m = mixer_of(inst, did);
  int b = inst >> 1;
  float v[4], s = 0.f, ss = 0.f;
#pragma unroll
  for (int j = 0; j < 4; ++j) {
    int c = lane + 32 * j;
    v[j] = x[((size_t)b * C_IN + c) * L_SEQ + l];
    s += v[j]; ss += v[j] * v[j];
  }
#pragma unroll
  for (int off = 16; off > 0; off >>= 1) {
    s  += __shfl_xor(s, off);
    ss += __shfl_xor(ss, off);
  }
  float mu = s * (1.f / 128.f);
  float rs = rsqrtf(ss * (1.f / 128.f) - mu * mu + 1e-5f);
  float* xn = ws + XN_OFF + ((size_t)inst * L_SEQ + l) * C_IN;
#pragma unroll
  for (int j = 0; j < 4; ++j) {
    int c = lane + 32 * j;
    xn[c] = (v[j] - mu) * rs * ln_g[m * C_IN + c] + ln_b[m * C_IN + c];
  }
}

// ---------------- kernel 2: in_proj  xz = xn_part @ in_w^T  (K=32) -----------
__global__ void k_inproj(float* __restrict__ ws, const float* __restrict__ in_w,
                         const int* __restrict__ did) {
  int wid  = (blockIdx.x << 3) + (threadIdx.x >> 5);
  int lane = threadIdx.x & 31;
  int nt = wid & 7;   wid >>= 3;
  int mt = wid & 255; wid >>= 8;
  int p  = wid & 3;   wid >>= 2;
  int inst = wid;
  int m = mixer_of(inst, did);
  const float* A = ws + XN_OFF + ((size_t)inst * L_SEQ + mt * 16) * C_IN + p * DM;
  const float* W = in_w + (size_t)m * (2 * DI) * DM + (size_t)(nt * 16) * DM;
  v8f c = {};
  c = wmma_f16(load_a16x32(A, C_IN, lane), load_b16x32(W, DM, 16, lane), c);
  float* D = ws + XZ_OFF +
             ((size_t)(inst * NPART + p) * L_SEQ + mt * 16) * C_IN + nt * 16;
  int n = lane & 15, mb = (lane >> 4) << 3;
#pragma unroll
  for (int i = 0; i < 8; ++i) D[(size_t)(mb + i) * C_IN + n] = c[i];
}

// ---------------- kernel 3: causal depthwise conv (k=4) + SiLU -> xi ---------
__global__ void k_conv(float* __restrict__ ws, const float* __restrict__ conv_w,
                       const float* __restrict__ conv_b,
                       const int* __restrict__ did) {
  int idx = blockIdx.x * 256 + threadIdx.x;
  int d  = idx & (DI - 1);
  int l  = (idx >> 6) & (L_SEQ - 1);
  int ip = idx >> 18;             // inst*4 + p
  int inst = ip >> 2;
  int m = mixer_of(inst, did);
  const float* xz = ws + XZ_OFF + (size_t)ip * L_SEQ * C_IN;
  float acc = conv_b[m * DI + d];
#pragma unroll
  for (int j = 0; j < 4; ++j) {
    int lj = l - 3 + j;
    float xv = (lj >= 0) ? xz[(size_t)lj * C_IN + d] : 0.f;
    acc += xv * conv_w[(m * DI + d) * 4 + j];
  }
  float sil = acc / (1.f + __expf(-acc));
  ws[XI_OFF + (size_t)ip * L_SEQ * DI + (size_t)l * DI + d] = sil;
}

// ---------------- kernel 4: x_proj  dbc = xi @ xp_w^T  (K=64, N=34->48) ------
__global__ void k_xproj(float* __restrict__ ws, const float* __restrict__ xp_w,
                        const int* __restrict__ did) {
  int wid  = (blockIdx.x << 3) + (threadIdx.x >> 5);
  int lane = threadIdx.x & 31;
  int nt = wid % 3;   wid /= 3;
  int mt = wid & 255; wid >>= 8;
  int p  = wid & 3;   wid >>= 2;
  int inst = wid;
  int m = mixer_of(inst, did);
  const float* A = ws + XI_OFF +
                   ((size_t)(inst * NPART + p) * L_SEQ + mt * 16) * DI;
  const float* W = xp_w + (size_t)m * 34 * DI + (size_t)(nt * 16) * DI;
  int nvalid = 34 - nt * 16; if (nvalid > 16) nvalid = 16;
  v8f c = {};
#pragma unroll
  for (int k0 = 0; k0 < DI; k0 += 32)
    c = wmma_f16(load_a16x32(A + k0, DI, lane),
                 load_b16x32(W + k0, DI, nvalid, lane), c);
  float* D = ws + DBC_OFF +
             ((size_t)(inst * NPART + p) * L_SEQ + mt * 16) * DBC_STRIDE + nt * 16;
  int n = lane & 15, mb = (lane >> 4) << 3;
#pragma unroll
  for (int i = 0; i < 8; ++i) D[(size_t)(mb + i) * DBC_STRIDE + n] = c[i];
}

// ---------------- kernel 5: delta = softplus(dt_low @ dt_w^T + dt_b) ---------
__global__ void k_delta(float* __restrict__ ws, const float* __restrict__ dt_w,
                        const float* __restrict__ dt_b,
                        const int* __restrict__ did) {
  int idx = blockIdx.x * 256 + threadIdx.x;
  int d  = idx & (DI - 1);
  int l  = (idx >> 6) & (L_SEQ - 1);
  int ip = idx >> 18;
  int inst = ip >> 2;
  int m = mixer_of(inst, did);
  const float* dbc = ws + DBC_OFF + ((size_t)ip * L_SEQ + l) * DBC_STRIDE;
  float t = dbc[0] * dt_w[(m * DI + d) * 2 + 0] +
            dbc[1] * dt_w[(m * DI + d) * 2 + 1] + dt_b[m * DI + d];
  float sp = (t > 20.f) ? t : log1pf(__expf(t));
  ws[DEL_OFF + (size_t)ip * L_SEQ * DI + (size_t)l * DI + d] = sp;
}

// ---------------- kernel 6: selective scan + D-skip + SiLU(z) gate -----------
// one workgroup (64 threads = 2 wave32) per (inst,part); thread d owns h[16].
__global__ void k_scan(float* __restrict__ ws, const float* __restrict__ A_log,
                       const float* __restrict__ Dp, const int* __restrict__ did) {
  int ip   = blockIdx.x;
  int inst = ip >> 2;
  int m = mixer_of(inst, did);
  int d = threadIdx.x;  // 0..63
  const float* del = ws + DEL_OFF + (size_t)ip * L_SEQ * DI;
  const float* xi  = ws + XI_OFF  + (size_t)ip * L_SEQ * DI;
  const float* dbc = ws + DBC_OFF + (size_t)ip * L_SEQ * DBC_STRIDE;
  const float* xz  = ws + XZ_OFF  + (size_t)ip * L_SEQ * C_IN;
  float* y = ws + Y_OFF + (size_t)ip * L_SEQ * DI;
  float aa[DSTATE], h[DSTATE];
  float dp = Dp[m * DI + d];
#pragma unroll
  for (int s = 0; s < DSTATE; ++s) {
    aa[s] = -__expf(A_log[(m * DI + d) * DSTATE + s]);
    h[s] = 0.f;
  }
  __shared__ float bc[32];  // [0..15]=B_t, [16..31]=C_t
  for (int t = 0; t < L_SEQ; ++t) {
    if (threadIdx.x < 32)
      bc[threadIdx.x] = dbc[(size_t)t * DBC_STRIDE + 2 + threadIdx.x];
    __syncthreads();
    float dv  = del[(size_t)t * DI + d];
    float xv  = xi[(size_t)t * DI + d];
    float dvx = dv * xv;
    float acc = 0.f;
#pragma unroll
    for (int s = 0; s < DSTATE; ++s) {
      float dA = __expf(dv * aa[s]);
      h[s] = dA * h[s] + dvx * bc[s];
      acc += h[s] * bc[16 + s];
    }
    float zz = xz[(size_t)t * C_IN + DI + d];
    float yv = (acc + dp * xv) * (zz / (1.f + __expf(-zz)));
    y[(size_t)t * DI + d] = yv;
    __syncthreads();
  }
}

// ---------------- kernel 7: out_proj + skip*part -> xm (inst,L,128) ----------
__global__ void k_outproj(float* __restrict__ ws, const float* __restrict__ out_w,
                          const float* __restrict__ skip,
                          const int* __restrict__ did) {
  int wid  = (blockIdx.x << 3) + (threadIdx.x >> 5);
  int lane = threadIdx.x & 31;
  int nt = wid & 1;   wid >>= 1;
  int mt = wid & 255; wid >>= 8;
  int p  = wid & 3;   wid >>= 2;
  int inst = wid;
  int m = mixer_of(inst, did);
  const float* A = ws + Y_OFF +
                   ((size_t)(inst * NPART + p) * L_SEQ + mt * 16) * DI;
  const float* W = out_w + (size_t)m * DM * DI + (size_t)(nt * 16) * DI;
  v8f c = {};
#pragma unroll
  for (int k0 = 0; k0 < DI; k0 += 32)
    c = wmma_f16(load_a16x32(A + k0, DI, lane),
                 load_b16x32(W + k0, DI, 16, lane), c);
  float sk = skip[m];
  int n = lane & 15, mb = (lane >> 4) << 3;
  size_t col = (size_t)(p * DM + nt * 16 + n);
  const float* xn = ws + XN_OFF + ((size_t)inst * L_SEQ + mt * 16) * C_IN + col;
  float* D        = ws + XM_OFF + ((size_t)inst * L_SEQ + mt * 16) * C_IN + col;
#pragma unroll
  for (int i = 0; i < 8; ++i)
    D[(size_t)(mb + i) * C_IN] = c[i] + sk * xn[(size_t)(mb + i) * C_IN];
}

// ---------------- kernel 8: layernorm of xm -> xm2 ---------------------------
__global__ void k_ln2(float* __restrict__ ws, const float* __restrict__ ln_g,
                      const float* __restrict__ ln_b,
                      const int* __restrict__ did) {
  int wid  = (blockIdx.x << 3) + (threadIdx.x >> 5);
  int lane = threadIdx.x & 31;
  int l    = wid & (L_SEQ - 1);
  int inst = wid >> 12;
  int m = mixer_of(inst, did);
  const float* xm = ws + XM_OFF + ((size_t)inst * L_SEQ + l) * C_IN;
  float v[4], s = 0.f, ss = 0.f;
#pragma unroll
  for (int j = 0; j < 4; ++j) {
    v[j] = xm[lane * 4 + j];
    s += v[j]; ss += v[j] * v[j];
  }
#pragma unroll
  for (int off = 16; off > 0; off >>= 1) {
    s  += __shfl_xor(s, off);
    ss += __shfl_xor(ss, off);
  }
  float mu = s * (1.f / 128.f);
  float rs = rsqrtf(ss * (1.f / 128.f) - mu * mu + 1e-5f);
  float* o = ws + XM2_OFF + ((size_t)inst * L_SEQ + l) * C_IN;
#pragma unroll
  for (int j = 0; j < 4; ++j) {
    int cidx = lane * 4 + j;
    o[cidx] = (v[j] - mu) * rs * ln_g[m * C_IN + cidx] + ln_b[m * C_IN + cidx];
  }
}

// ---------------- kernel 9: final proj  pj = xm2 @ proj_w^T  (K=128) ---------
__global__ void k_proj(float* __restrict__ ws, const float* __restrict__ proj_w,
                       const int* __restrict__ did) {
  int wid  = (blockIdx.x << 3) + (threadIdx.x >> 5);
  int lane = threadIdx.x & 31;
  int nt = wid & 7;   wid >>= 3;
  int mt = wid & 255; wid >>= 8;
  int inst = wid;
  int m = mixer_of(inst, did);
  const float* A = ws + XM2_OFF + ((size_t)inst * L_SEQ + mt * 16) * C_IN;
  const float* W = proj_w + (size_t)m * C_IN * C_IN + (size_t)(nt * 16) * C_IN;
  v8f c = {};
#pragma unroll
  for (int k0 = 0; k0 < C_IN; k0 += 32)
    c = wmma_f16(load_a16x32(A + k0, C_IN, lane),
                 load_b16x32(W + k0, C_IN, 16, lane), c);
  float* D = ws + PJ_OFF + ((size_t)inst * L_SEQ + mt * 16) * C_IN + nt * 16;
  int n = lane & 15, mb = (lane >> 4) << 3;
#pragma unroll
  for (int i = 0; i < 8; ++i) D[(size_t)(mb + i) * C_IN + n] = c[i];
}

// ---------------- kernel 10: shared + expert + biases, transpose to output ---
__global__ void k_combine(const float* __restrict__ ws,
                          const float* __restrict__ proj_b,
                          const int* __restrict__ did, float* __restrict__ out) {
  int idx = blockIdx.x * 256 + threadIdx.x;  // = b*128*4096 + o*4096 + l
  int l = idx & (L_SEQ - 1);
  int o = (idx >> 12) & 127;
  int b = idx >> 19;
  int me = mixer_of(2 * b + 1, did);
  const float* pj = ws + PJ_OFF;
  float v = pj[((size_t)(2 * b)     * L_SEQ + l) * C_IN + o] + proj_b[o] +
            pj[((size_t)(2 * b + 1) * L_SEQ + l) * C_IN + o] +
            proj_b[me * C_IN + o];
  out[idx] = v;
}

extern "C" void kernel_launch(void* const* d_in, const int* in_sizes, int n_in,
                              void* d_out, int out_size, void* d_ws,
                              size_t ws_size, hipStream_t stream) {
  const float* x      = (const float*)d_in[0];
  const int*   did    = (const int*)  d_in[1];
  const float* ln_g   = (const float*)d_in[2];
  const float* ln_b   = (const float*)d_in[3];
  const float* in_w   = (const float*)d_in[4];
  const float* conv_w = (const float*)d_in[5];
  const float* conv_b = (const float*)d_in[6];
  const float* xp_w   = (const float*)d_in[7];
  const float* dt_w   = (const float*)d_in[8];
  const float* dt_b   = (const float*)d_in[9];
  const float* A_log  = (const float*)d_in[10];
  const float* Dp     = (const float*)d_in[11];
  const float* out_w  = (const float*)d_in[12];
  const float* proj_w = (const float*)d_in[13];
  const float* proj_b = (const float*)d_in[14];
  const float* skip   = (const float*)d_in[15];
  float* ws  = (float*)d_ws;
  float* out = (float*)d_out;
  (void)in_sizes; (void)n_in; (void)out_size; (void)ws_size;

  k_ln1    <<<2048,  256, 0, stream>>>(x, ln_g, ln_b, did, ws);
  k_inproj <<<4096,  256, 0, stream>>>(ws, in_w, did);
  k_conv   <<<16384, 256, 0, stream>>>(ws, conv_w, conv_b, did);
  k_xproj  <<<1536,  256, 0, stream>>>(ws, xp_w, did);
  k_delta  <<<16384, 256, 0, stream>>>(ws, dt_w, dt_b, did);
  k_scan   <<<16,    64,  0, stream>>>(ws, A_log, Dp, did);
  k_outproj<<<1024,  256, 0, stream>>>(ws, out_w, skip, did);
  k_ln2    <<<2048,  256, 0, stream>>>(ws, ln_g, ln_b, did);
  k_proj   <<<1024,  256, 0, stream>>>(ws, proj_w, did);
  k_combine<<<4096,  256, 0, stream>>>(ws, proj_b, did, out);
}
